// UnnamedModel_45526653337613
// MI455X (gfx1250) — matrix operand order repaired and checked by reading, not compiled, not run on hardware
//
#include <hip/hip_runtime.h>

// ---------------------------------------------------------------------------
// CDNA5 (gfx1250) hetero-GAT forward.
//  * all GEMMs: v_wmma_f32_16x16x32_bf16, bf16 operands pre-staged in global
//  * tile staging: global_load_async_to_lds_b32 + s_wait_asynccnt (ASYNCcnt DMA)
//  * weights pre-transposed to [N][K] so A and B fragments are contiguous-K
//  * edge softmax/scatter reads bf16 projections (halves the ~3.5GB gather)
// ---------------------------------------------------------------------------

typedef __attribute__((ext_vector_type(16))) __bf16 v16bf;
typedef __attribute__((ext_vector_type(8)))  float  v8f;

__device__ __forceinline__ unsigned short f2bf(float f) {
    unsigned int u = __float_as_uint(f);
    unsigned int r = u + 0x7FFFu + ((u >> 16) & 1u);   // round-to-nearest-even
    return (unsigned short)(r >> 16);
}
__device__ __forceinline__ float bf2f(unsigned short h) {
    return __uint_as_float(((unsigned int)h) << 16);
}

// gfx1250 async DMA: global -> LDS, tracked by ASYNCcnt
__device__ __forceinline__ void async_load_b32(unsigned lds_addr, const void* gaddr) {
    asm volatile("global_load_async_to_lds_b32 %0, %1, off"
                 :: "v"(lds_addr), "v"((unsigned long long)(size_t)gaddr)
                 : "memory");
}
__device__ __forceinline__ void wait_async0() {
    asm volatile("s_wait_asynccnt 0" ::: "memory");
}

// ---------------------------------------------------------------------------
// GEMM: C[M,N] = A[M,K] x B[K,N], with A bf16 [M][K], Bt bf16 [N][K].
// Block 256 threads (8 waves), tile 128x64, K-step 32.
// Wave grid 4(M) x 2(N), each wave owns 32x32 = 2x2 wmma accumulators.
// ---------------------------------------------------------------------------
#define TILE_M 128
#define TILE_N 64
#define TILE_K 32
#define LDS_S  40          // padded bf16 stride: 80B rows, 16B aligned, bank-friendly

__global__ __launch_bounds__(256)
void gemm_bf16_wmma(const unsigned short* __restrict__ A,
                    const unsigned short* __restrict__ Bt,
                    const float* __restrict__ bias,
                    float* __restrict__ Cf, unsigned short* __restrict__ Cb,
                    int M, int N, int K, int do_relu)
{
    __shared__ unsigned short Asm[TILE_M * LDS_S];
    __shared__ unsigned short Bsm[TILE_N * LDS_S];

    const int tid  = threadIdx.x;
    const int lane = tid & 31;
    const int wave = tid >> 5;
    const int wm   = wave & 3;
    const int wn   = wave >> 2;
    const int row0 = blockIdx.y * TILE_M;
    const int col0 = blockIdx.x * TILE_N;

    const unsigned asmBase = (unsigned)(size_t)&Asm[0];
    const unsigned bsmBase = (unsigned)(size_t)&Bsm[0];

    v8f acc[2][2];
    #pragma unroll
    for (int i = 0; i < 2; ++i)
        #pragma unroll
        for (int j = 0; j < 2; ++j)
            #pragma unroll
            for (int e = 0; e < 8; ++e) acc[i][j][e] = 0.0f;

    const int rsel   = lane & 15;
    const int hi     = lane >> 4;
    const int khalfA = hi * 16;
    const int kbaseB = hi * 16;

    union U16 { v16bf v; unsigned int u[8]; };

    const bool fullMN = (row0 + TILE_M <= M) && (col0 + TILE_N <= N);
    const int nkt = (K + TILE_K - 1) / TILE_K;

    for (int kt = 0; kt < nkt; ++kt) {
        const int k0 = kt * TILE_K;

        if (fullMN && (k0 + TILE_K <= K)) {
            // ---------- interior fast path: async DMA into LDS ----------
            // A tile: 128 rows x 64B = 2048 dwords; 8 per thread
            #pragma unroll
            for (int i = 0; i < 8; ++i) {
                int id = tid + i * 256;
                int r = id >> 4, c = id & 15;
                const char* g = (const char*)A + ((size_t)(row0 + r) * K + k0) * 2 + c * 4;
                async_load_b32(asmBase + (unsigned)(r * (LDS_S * 2) + c * 4), g);
            }
            // B tile: 64 rows x 64B = 1024 dwords; 4 per thread
            #pragma unroll
            for (int i = 0; i < 4; ++i) {
                int id = tid + i * 256;
                int r = id >> 4, c = id & 15;
                const char* g = (const char*)Bt + ((size_t)(col0 + r) * K + k0) * 2 + c * 4;
                async_load_b32(bsmBase + (unsigned)(r * (LDS_S * 2) + c * 4), g);
            }
            // L2 prefetch of next K tile while DMA runs
            if (kt + 1 < nkt) {
                int nk = k0 + TILE_K;
                if (nk < K) {
                    __builtin_prefetch((const char*)A + ((size_t)(row0 + (tid >> 1)) * K + nk) * 2, 0, 1);
                    __builtin_prefetch((const char*)Bt + ((size_t)(col0 + (tid & 63)) * K + nk) * 2, 0, 1);
                }
            }
            wait_async0();
        } else {
            // ---------- edge path: branch-free clamped loads ----------
            #pragma unroll
            for (int i = 0; i < 16; ++i) {
                int lin = tid + i * 256;
                int r = lin >> 5, c = lin & 31;
                int gr = row0 + r, gk = k0 + c;
                int cr = gr < M ? gr : M - 1;
                int ck = gk < K ? gk : K - 1;
                unsigned short v = A[(size_t)cr * K + ck];
                if (gr >= M || gk >= K) v = 0;
                Asm[r * LDS_S + c] = v;
            }
            #pragma unroll
            for (int i = 0; i < 8; ++i) {
                int lin = tid + i * 256;
                int r = lin >> 5, c = lin & 31;
                int gn = col0 + r, gk = k0 + c;
                int cn = gn < N ? gn : N - 1;
                int ck = gk < K ? gk : K - 1;
                unsigned short v = Bt[(size_t)cn * K + ck];
                if (gn >= N || gk >= K) v = 0;
                Bsm[r * LDS_S + c] = v;
            }
        }
        __syncthreads();

        // ---------- fragments (contiguous K: element s holds k = 16*hi + s) ----------
        U16 afrag[2], bfrag[2];
        #pragma unroll
        for (int ti = 0; ti < 2; ++ti) {
            int r = wm * 32 + ti * 16 + rsel;
            const unsigned short* pa = &Asm[r * LDS_S + khalfA];
            #pragma unroll
            for (int j = 0; j < 4; ++j) {
                afrag[ti].u[j]     = *(const unsigned int*)(pa + 2 * j);
                afrag[ti].u[4 + j] = *(const unsigned int*)(pa + 8 + 2 * j);
            }
        }
        #pragma unroll
        for (int tj = 0; tj < 2; ++tj) {
            int ncol = wn * 32 + tj * 16 + rsel;
            const unsigned short* pb = &Bsm[ncol * LDS_S + kbaseB];
            #pragma unroll
            for (int j = 0; j < 8; ++j)
                bfrag[tj].u[j] = *(const unsigned int*)(pb + 2 * j);
        }

        #pragma unroll
        for (int ti = 0; ti < 2; ++ti)
            #pragma unroll
            for (int tj = 0; tj < 2; ++tj)
                acc[ti][tj] = __builtin_amdgcn_wmma_f32_16x16x32_bf16(
                    false, afrag[ti].v, false, bfrag[tj].v,
                    (short)0, acc[ti][tj], false, false);

        __syncthreads();
    }

    // ---------- epilogue: D layout M = r + 8*hi, N = lane%16 ----------
    #pragma unroll
    for (int ti = 0; ti < 2; ++ti)
        #pragma unroll
        for (int tj = 0; tj < 2; ++tj) {
            int col = col0 + wn * 32 + tj * 16 + rsel;
            if (col >= N) continue;
            float bv = bias ? bias[col] : 0.0f;
            #pragma unroll
            for (int r = 0; r < 8; ++r) {
                int grow = row0 + wm * 32 + ti * 16 + r + 8 * hi;
                if (grow < M) {
                    float v = acc[ti][tj][r] + bv;
                    if (do_relu) v = fmaxf(v, 0.0f);
                    if (Cb) Cb[(size_t)grow * N + col] = f2bf(v);
                    else    Cf[(size_t)grow * N + col] = v;
                }
            }
        }
}

// fp32 -> bf16 flat copy
__global__ void convert_bf16_kernel(const float* __restrict__ x,
                                    unsigned short* __restrict__ y, size_t n)
{
    size_t i = (size_t)blockIdx.x * blockDim.x + threadIdx.x;
    if (i < n) y[i] = f2bf(x[i]);
}

// W[K][N] fp32 -> Wt[N][K] bf16 (tiled transpose)
__global__ void transpose_bf16_kernel(const float* __restrict__ W,
                                      unsigned short* __restrict__ Wt, int K, int N)
{
    __shared__ float tile[32][33];
    int k0 = blockIdx.y * 32, n0 = blockIdx.x * 32;
    int tx = threadIdx.x & 31, ty = threadIdx.x >> 5;  // 256 thr: ty 0..7
    for (int i = ty; i < 32; i += 8) {
        int k = k0 + i, n = n0 + tx;
        tile[i][tx] = (k < K && n < N) ? W[(size_t)k * N + n] : 0.0f;
    }
    __syncthreads();
    for (int i = ty; i < 32; i += 8) {
        int n = n0 + i, k = k0 + tx;
        if (n < N && k < K) Wt[(size_t)n * K + k] = f2bf(tile[tx][i]);
    }
}

// s[row] = dot(bf16 H[row], fp32 a) — one wave32 per row
__global__ void rowdot_bf16_kernel(const unsigned short* __restrict__ H,
                                   const float* __restrict__ a,
                                   float* __restrict__ s, int nrows, int dim)
{
    int row  = blockIdx.x * (blockDim.x >> 5) + (threadIdx.x >> 5);
    int lane = threadIdx.x & 31;
    if (row >= nrows) return;
    float acc = 0.0f;
    for (int c = lane; c < dim; c += 32) acc += bf2f(H[(size_t)row * dim + c]) * a[c];
    #pragma unroll
    for (int o = 16; o > 0; o >>= 1) acc += __shfl_xor(acc, o, 32);
    if (lane == 0) s[row] = acc;
}

__global__ void fill_kernel(float* __restrict__ p, float v, size_t n)
{
    size_t i = (size_t)blockIdx.x * blockDim.x + threadIdx.x;
    if (i < n) p[i] = v;
}

__device__ __forceinline__ void atomicMaxFloat(float* addr, float val)
{
    unsigned int* ua  = (unsigned int*)addr;
    unsigned int  old = *ua;
    while (true) {
        float f = __uint_as_float(old);
        if (f >= val) break;
        unsigned int assumed = old;
        old = atomicCAS(ua, assumed, __float_as_uint(val));
        if (old == assumed) break;
    }
}

__global__ void edge_att_kernel(const int* __restrict__ src, const int* __restrict__ dst,
                                int E, int nself,
                                const float* __restrict__ sS, const float* __restrict__ sD,
                                float* __restrict__ e_out, float* __restrict__ emax)
{
    int i = blockIdx.x * blockDim.x + threadIdx.x;
    if (i >= E + nself) return;
    int s = (i < E) ? src[i] : (i - E);
    int d = (i < E) ? dst[i] : (i - E);
    float e = sS[s] + sD[d];
    e = (e > 0.0f) ? e : 0.2f * e;
    e_out[i] = e;
    atomicMaxFloat(&emax[d], e);
}

__global__ void edge_exp_kernel(const int* __restrict__ src, const int* __restrict__ dst,
                                int E, int nself,
                                float* __restrict__ w, const float* __restrict__ emax,
                                float* __restrict__ denom)
{
    int i = blockIdx.x * blockDim.x + threadIdx.x;
    if (i >= E + nself) return;
    int d = (i < E) ? dst[i] : (i - E);
    float wv = __expf(w[i] - emax[d]);
    w[i] = wv;
    atomicAdd(&denom[d], wv);
}

// out[dst] += bf16 hs[src] * alpha — one block per edge, strided channels
__global__ void edge_scatter_kernel(const int* __restrict__ src, const int* __restrict__ dst,
                                    int E,
                                    const float* __restrict__ w, const float* __restrict__ denom,
                                    const unsigned short* __restrict__ hs,
                                    float* __restrict__ out, int Cc)
{
    int i = blockIdx.x;
    int s = (i < E) ? src[i] : (i - E);
    int d = (i < E) ? dst[i] : (i - E);
    float alpha = w[i] / (denom[d] + 1e-16f);
    const unsigned short* hrow = hs + (size_t)s * Cc;
    float*                orow = out + (size_t)d * Cc;
    for (int c = threadIdx.x; c < Cc; c += blockDim.x)
        atomicAdd(&orow[c], bf2f(hrow[c]) * alpha);
}

__global__ void combine_relu_kernel(const float* __restrict__ a, const float* __restrict__ b,
                                    const float* __restrict__ b1, const float* __restrict__ b2,
                                    float* __restrict__ out, size_t total, int Cc)
{
    size_t i = (size_t)blockIdx.x * blockDim.x + threadIdx.x;
    if (i >= total) return;
    int c = (int)(i % Cc);
    out[i] = fmaxf(a[i] + b[i] + b1[c] + b2[c], 0.0f);
}

// row-wise L2 normalize, fp32 in -> bf16 out (GEMM operand)
__global__ void l2norm_bf16_kernel(const float* __restrict__ x,
                                   unsigned short* __restrict__ y, int dim)
{
    __shared__ float red[8];
    int row = blockIdx.x;
    float acc = 0.0f;
    for (int c = threadIdx.x; c < dim; c += blockDim.x) {
        float v = x[(size_t)row * dim + c];
        acc += v * v;
    }
    #pragma unroll
    for (int o = 16; o > 0; o >>= 1) acc += __shfl_xor(acc, o, 32);
    int lane = threadIdx.x & 31, wv = threadIdx.x >> 5;
    if (lane == 0) red[wv] = acc;
    __syncthreads();
    if (threadIdx.x == 0) {
        float t = 0.0f;
        for (int j = 0; j < (int)(blockDim.x >> 5); ++j) t += red[j];
        red[0] = t;
    }
    __syncthreads();
    float inv = 1.0f / fmaxf(sqrtf(red[0]), 1e-12f);
    for (int c = threadIdx.x; c < dim; c += blockDim.x)
        y[(size_t)row * dim + c] = f2bf(x[(size_t)row * dim + c] * inv);
}

// hidden[row] = [drug_h[d1], drug_h[d2], cell[row]] (768+768+1536)
__global__ void concat_kernel(const int* __restrict__ d1, const int* __restrict__ d2,
                              const float* __restrict__ drug_h, const float* __restrict__ cell,
                              float* __restrict__ hidden)
{
    int row = blockIdx.x;
    int i1 = d1[row], i2 = d2[row];
    for (int c = threadIdx.x; c < 3072; c += blockDim.x) {
        float v;
        if (c < 768)       v = drug_h[(size_t)i1 * 768 + c];
        else if (c < 1536) v = drug_h[(size_t)i2 * 768 + (c - 768)];
        else               v = cell[(size_t)row * 1536 + (c - 1536)];
        hidden[(size_t)row * 3072 + c] = v;
    }
}

__global__ void cls_softmax_kernel(const float* __restrict__ h, const float* __restrict__ W,
                                   const float* __restrict__ b, float* __restrict__ out, int Bn)
{
    int row  = blockIdx.x * (blockDim.x >> 5) + (threadIdx.x >> 5);
    int lane = threadIdx.x & 31;
    if (row >= Bn) return;
    float a0 = 0.0f, a1 = 0.0f;
    for (int c = lane; c < 768; c += 32) {
        float hv = h[(size_t)row * 768 + c];
        a0 += hv * W[c * 2 + 0];
        a1 += hv * W[c * 2 + 1];
    }
    #pragma unroll
    for (int o = 16; o > 0; o >>= 1) { a0 += __shfl_xor(a0, o, 32); a1 += __shfl_xor(a1, o, 32); }
    if (lane == 0) {
        a0 += b[0]; a1 += b[1];
        float m = fmaxf(a0, a1);
        float e0 = __expf(a0 - m), e1 = __expf(a1 - m);
        float inv = 1.0f / (e0 + e1);
        out[row * 2 + 0] = e0 * inv;
        out[row * 2 + 1] = e1 * inv;
    }
}

// ---------------------------------------------------------------------------
extern "C" void kernel_launch(void* const* d_in, const int* in_sizes, int n_in,
                              void* d_out, int out_size, void* d_ws, size_t ws_size,
                              hipStream_t stream)
{
    (void)in_sizes; (void)n_in; (void)out_size; (void)ws_size;

    const int ND = 3000, NT = 10000, FD = 200, FT = 570, Cc = 768, Bn = 4096;
    const int E_DD = 100000, E_DT = 150000, E_RDT = 150000, E_TT = 150000;
    typedef unsigned short u16;

    const int*   drug1 = (const int*)d_in[0];
    const int*   drug2 = (const int*)d_in[1];
    const float* cellf = (const float*)d_in[2];
    const float* xd    = (const float*)d_in[3];
    const float* xt    = (const float*)d_in[4];
    const int *dd_src = (const int*)d_in[5],  *dd_dst = (const int*)d_in[6];
    const int *dt_src = (const int*)d_in[7],  *dt_dst = (const int*)d_in[8];
    const int *rdt_src = (const int*)d_in[9], *rdt_dst = (const int*)d_in[10];
    const int *tt_src = (const int*)d_in[11], *tt_dst = (const int*)d_in[12];
    const float* clsW = (const float*)d_in[13]; const float* clsb = (const float*)d_in[14];
    const float* ddWd = (const float*)d_in[15]; const float* ddWs = (const float*)d_in[16];
    const float* ddad = (const float*)d_in[17]; const float* ddas = (const float*)d_in[18];
    const float* ddb  = (const float*)d_in[19];
    const float* dtWd = (const float*)d_in[20]; const float* dtWs = (const float*)d_in[21];
    const float* dtad = (const float*)d_in[22]; const float* dtas = (const float*)d_in[23];
    const float* dtb  = (const float*)d_in[24];
    const float* r1W1 = (const float*)d_in[25]; const float* r1W2 = (const float*)d_in[26];
    const float* r1W3 = (const float*)d_in[27];
    const float* r1b1 = (const float*)d_in[28]; const float* r1b2 = (const float*)d_in[29];
    const float* r1b3 = (const float*)d_in[30];
    const float* r2W1 = (const float*)d_in[31]; const float* r2W2 = (const float*)d_in[32];
    const float* r2W3 = (const float*)d_in[33];
    const float* r2b1 = (const float*)d_in[34]; const float* r2b2 = (const float*)d_in[35];
    const float* r2b3 = (const float*)d_in[36];
    const float* rdtWd = (const float*)d_in[37]; const float* rdtWs = (const float*)d_in[38];
    const float* rdtad = (const float*)d_in[39]; const float* rdtas = (const float*)d_in[40];
    const float* rdtb  = (const float*)d_in[41];
    const float* ttWd = (const float*)d_in[42]; const float* ttWs = (const float*)d_in[43];
    const float* ttad = (const float*)d_in[44]; const float* ttas = (const float*)d_in[45];
    const float* ttb  = (const float*)d_in[46];

    float* outp   = (float*)d_out;
    float* probs  = outp;
    float* drug_h = outp + (size_t)Bn * 2;
    float* targ_h = drug_h + (size_t)ND * Cc;

    char* wsb = (char*)d_ws;
    size_t off = 0;
    auto alloc = [&](size_t bytes) {
        void* p = wsb + off;
        off += (bytes + 255) & ~(size_t)255;
        return p;
    };

    auto gemm = [&](const u16* A, const u16* Bt, const float* bias,
                    float* Cf, u16* Cb, int M, int N, int K, int relu) {
        dim3 grid((N + TILE_N - 1) / TILE_N, (M + TILE_M - 1) / TILE_M);
        gemm_bf16_wmma<<<grid, dim3(256), 0, stream>>>(A, Bt, bias, Cf, Cb, M, N, K, relu);
    };
    auto convert = [&](const float* x, u16* y, size_t n) {
        convert_bf16_kernel<<<(unsigned)((n + 255) / 256), 256, 0, stream>>>(x, y, n);
    };
    auto transW = [&](const float* W, u16* Wt, int K, int N) {
        dim3 grid((N + 31) / 32, (K + 31) / 32);
        transpose_bf16_kernel<<<grid, 256, 0, stream>>>(W, Wt, K, N);
    };
    auto fill = [&](float* p, float v, size_t n) {
        fill_kernel<<<(unsigned)((n + 255) / 256), 256, 0, stream>>>(p, v, n);
    };
    auto rowdot = [&](const u16* H, const float* a, float* s, int rows) {
        rowdot_bf16_kernel<<<(rows + 3) / 4, 128, 0, stream>>>(H, a, s, rows, Cc);
    };

    // ============ Phase 0: persistent bf16 operands (activations + W^T) ====
    u16* xd_bf = (u16*)alloc((size_t)ND * FD * 2);
    u16* xt_bf = (u16*)alloc((size_t)NT * FT * 2);
    u16* ddWst = (u16*)alloc((size_t)Cc * FD * 2);
    u16* ddWdt = (u16*)alloc((size_t)Cc * FD * 2);
    u16* dtWst = (u16*)alloc((size_t)Cc * FD * 2);
    u16* dtWdt = (u16*)alloc((size_t)Cc * FT * 2);
    u16* rdtWst = (u16*)alloc((size_t)Cc * FT * 2);
    u16* rdtWdt = (u16*)alloc((size_t)Cc * FD * 2);
    u16* ttWst = (u16*)alloc((size_t)Cc * FT * 2);
    u16* ttWdt = (u16*)alloc((size_t)Cc * FT * 2);
    u16* r1W1t = (u16*)alloc((size_t)2048 * 890 * 2);
    u16* r1W2t = (u16*)alloc((size_t)512 * 2048 * 2);
    u16* r1W3t = (u16*)alloc((size_t)1536 * 512 * 2);
    u16* r2W1t = (u16*)alloc((size_t)2048 * 3072 * 2);
    u16* r2W2t = (u16*)alloc((size_t)512 * 2048 * 2);
    u16* r2W3t = (u16*)alloc((size_t)768 * 512 * 2);
    const size_t persist_end = off;

    convert(xd, xd_bf, (size_t)ND * FD);
    convert(xt, xt_bf, (size_t)NT * FT);
    transW(ddWs, ddWst, FD, Cc);   transW(ddWd, ddWdt, FD, Cc);
    transW(dtWs, dtWst, FD, Cc);   transW(dtWd, dtWdt, FT, Cc);
    transW(rdtWs, rdtWst, FT, Cc); transW(rdtWd, rdtWdt, FD, Cc);
    transW(ttWs, ttWst, FT, Cc);   transW(ttWd, ttWdt, FT, Cc);
    transW(r1W1, r1W1t, 890, 2048);
    transW(r1W2, r1W2t, 2048, 512);
    transW(r1W3, r1W3t, 512, 1536);
    transW(r2W1, r2W1t, 3072, 2048);
    transW(r2W2, r2W2t, 2048, 512);
    transW(r2W3, r2W3t, 512, 768);

    // ===================== Phase A: GAT layer ==============================
    u16* hs_dd  = (u16*)alloc((size_t)ND * Cc * 2);
    u16* hd_dd  = (u16*)alloc((size_t)ND * Cc * 2);
    u16* hs_rdt = (u16*)alloc((size_t)NT * Cc * 2);
    u16* hd_rdt = (u16*)alloc((size_t)ND * Cc * 2);
    u16* hs_dt  = (u16*)alloc((size_t)ND * Cc * 2);
    u16* hd_dt  = (u16*)alloc((size_t)NT * Cc * 2);
    u16* hs_tt  = (u16*)alloc((size_t)NT * Cc * 2);
    u16* hd_tt  = (u16*)alloc((size_t)NT * Cc * 2);
    float* out_dd  = (float*)alloc((size_t)ND * Cc * 4);
    float* out_rdt = (float*)alloc((size_t)ND * Cc * 4);
    float* out_dt  = (float*)alloc((size_t)NT * Cc * 4);
    float* out_tt  = (float*)alloc((size_t)NT * Cc * 4);
    float* sS_dd = (float*)alloc(ND * 4);  float* sD_dd = (float*)alloc(ND * 4);
    float* sS_rdt = (float*)alloc(NT * 4); float* sD_rdt = (float*)alloc(ND * 4);
    float* sS_dt = (float*)alloc(ND * 4);  float* sD_dt = (float*)alloc(NT * 4);
    float* sS_tt = (float*)alloc(NT * 4);  float* sD_tt = (float*)alloc(NT * 4);
    float* mx_dd = (float*)alloc(ND * 4);  float* dn_dd = (float*)alloc(ND * 4);
    float* mx_rdt = (float*)alloc(ND * 4); float* dn_rdt = (float*)alloc(ND * 4);
    float* mx_dt = (float*)alloc(NT * 4);  float* dn_dt = (float*)alloc(NT * 4);
    float* mx_tt = (float*)alloc(NT * 4);  float* dn_tt = (float*)alloc(NT * 4);
    float* w_dd  = (float*)alloc(((size_t)E_DD + ND) * 4);
    float* w_dt  = (float*)alloc((size_t)E_DT * 4);
    float* w_rdt = (float*)alloc((size_t)E_RDT * 4);
    float* w_tt  = (float*)alloc(((size_t)E_TT + NT) * 4);

    // projections (WMMA, bf16 out for rowdot + edge gather)
    gemm(xd_bf, ddWst,  nullptr, nullptr, hs_dd,  ND, Cc, FD, 0);
    gemm(xd_bf, ddWdt,  nullptr, nullptr, hd_dd,  ND, Cc, FD, 0);
    gemm(xt_bf, rdtWst, nullptr, nullptr, hs_rdt, NT, Cc, FT, 0);
    gemm(xd_bf, rdtWdt, nullptr, nullptr, hd_rdt, ND, Cc, FD, 0);
    gemm(xd_bf, dtWst,  nullptr, nullptr, hs_dt,  ND, Cc, FD, 0);
    gemm(xt_bf, dtWdt,  nullptr, nullptr, hd_dt,  NT, Cc, FT, 0);
    gemm(xt_bf, ttWst,  nullptr, nullptr, hs_tt,  NT, Cc, FT, 0);
    gemm(xt_bf, ttWdt,  nullptr, nullptr, hd_tt,  NT, Cc, FT, 0);

    rowdot(hs_dd, ddas, sS_dd, ND);    rowdot(hd_dd, ddad, sD_dd, ND);
    rowdot(hs_rdt, rdtas, sS_rdt, NT); rowdot(hd_rdt, rdtad, sD_rdt, ND);
    rowdot(hs_dt, dtas, sS_dt, ND);    rowdot(hd_dt, dtad, sD_dt, NT);
    rowdot(hs_tt, ttas, sS_tt, NT);    rowdot(hd_tt, ttad, sD_tt, NT);

    const float NEG = -3.402823466e38f;
    fill(mx_dd, NEG, ND);  fill(dn_dd, 0.f, ND);
    fill(mx_rdt, NEG, ND); fill(dn_rdt, 0.f, ND);
    fill(mx_dt, NEG, NT);  fill(dn_dt, 0.f, NT);
    fill(mx_tt, NEG, NT);  fill(dn_tt, 0.f, NT);
    fill(out_dd, 0.f, (size_t)ND * Cc);  fill(out_rdt, 0.f, (size_t)ND * Cc);
    fill(out_dt, 0.f, (size_t)NT * Cc);  fill(out_tt, 0.f, (size_t)NT * Cc);

    auto edge_type = [&](const int* src, const int* dst, int E, int nself,
                         float* wbuf, float* mx, float* dn, const float* sS,
                         const float* sD, const u16* hs, float* outb) {
        int tot = E + nself;
        int g = (tot + 255) / 256;
        edge_att_kernel<<<g, 256, 0, stream>>>(src, dst, E, nself, sS, sD, wbuf, mx);
        edge_exp_kernel<<<g, 256, 0, stream>>>(src, dst, E, nself, wbuf, mx, dn);
        edge_scatter_kernel<<<tot, 256, 0, stream>>>(src, dst, E, wbuf, dn, hs, outb, Cc);
    };
    edge_type(dd_src, dd_dst, E_DD, ND, w_dd, mx_dd, dn_dd, sS_dd, sD_dd, hs_dd, out_dd);
    edge_type(rdt_src, rdt_dst, E_RDT, 0, w_rdt, mx_rdt, dn_rdt, sS_rdt, sD_rdt, hs_rdt, out_rdt);
    edge_type(dt_src, dt_dst, E_DT, 0, w_dt, mx_dt, dn_dt, sS_dt, sD_dt, hs_dt, out_dt);
    edge_type(tt_src, tt_dst, E_TT, NT, w_tt, mx_tt, dn_tt, sS_tt, sD_tt, hs_tt, out_tt);

    {
        size_t td = (size_t)ND * Cc, tt = (size_t)NT * Cc;
        combine_relu_kernel<<<(unsigned)((td + 255) / 256), 256, 0, stream>>>(
            out_dd, out_rdt, ddb, rdtb, drug_h, td, Cc);
        combine_relu_kernel<<<(unsigned)((tt + 255) / 256), 256, 0, stream>>>(
            out_dt, out_tt, dtb, ttb, targ_h, tt, Cc);
    }

    // ===================== Phase B: MLP head (reuses phase-A scratch) ======
    off = persist_end;
    u16*   n1_bf   = (u16*)alloc((size_t)Bn * 890 * 2);
    u16*   c1_bf   = (u16*)alloc((size_t)Bn * 2048 * 2);
    u16*   c2_bf   = (u16*)alloc((size_t)Bn * 512 * 2);
    float* cell    = (float*)alloc((size_t)Bn * 1536 * 4);
    float* hidden  = (float*)alloc((size_t)Bn * 3072 * 4);
    u16*   hid_bf  = (u16*)alloc((size_t)Bn * 3072 * 2);
    u16*   g1_bf   = (u16*)alloc((size_t)Bn * 2048 * 2);
    u16*   g2_bf   = (u16*)alloc((size_t)Bn * 512 * 2);
    float* hfin    = (float*)alloc((size_t)Bn * 768 * 4);

    l2norm_bf16_kernel<<<Bn, 256, 0, stream>>>(cellf, n1_bf, 890);
    gemm(n1_bf, r1W1t, r1b1, nullptr, c1_bf, Bn, 2048, 890, 1);
    gemm(c1_bf, r1W2t, r1b2, nullptr, c2_bf, Bn, 512, 2048, 1);
    gemm(c2_bf, r1W3t, r1b3, cell, nullptr, Bn, 1536, 512, 1);

    concat_kernel<<<Bn, 256, 0, stream>>>(drug1, drug2, drug_h, cell, hidden);
    l2norm_bf16_kernel<<<Bn, 256, 0, stream>>>(hidden, hid_bf, 3072);

    gemm(hid_bf, r2W1t, r2b1, nullptr, g1_bf, Bn, 2048, 3072, 1);
    gemm(g1_bf, r2W2t, r2b2, nullptr, g2_bf, Bn, 512, 2048, 1);
    gemm(g2_bf, r2W3t, r2b3, hfin, nullptr, Bn, 768, 512, 1);

    cls_softmax_kernel<<<(Bn + 3) / 4, 128, 0, stream>>>(hfin, clsW, clsb, probs, Bn);
}
